// Transformer_Encoder_1941325218503
// MI455X (gfx1250) — compile-verified, matrix operand order checked by reading, not run on hardware
//
#include <hip/hip_runtime.h>

// MI455X / gfx1250 wave32 WMMA implementation of the per-head attention encoder.
// Pipeline: proj(Q) proj(K) proj(V^T) -> flash attention -> combine GEMM.
// GEMMs use v_wmma_f32_16x16x32_f16; K/V streaming uses double-buffered
// global_load_async_to_lds_b128 (ASYNCcnt) with one barrier per chunk.

#define BATCH   2048
#define HEADS   32
#define DHEAD   64
#define DMODEL  2048

typedef __attribute__((ext_vector_type(16))) _Float16 v16h;
typedef __attribute__((ext_vector_type(8)))  _Float16 v8h;
typedef __attribute__((ext_vector_type(8)))  float    v8f;
typedef __attribute__((ext_vector_type(4)))  float    v4f;

__device__ __forceinline__ v8f wmma16(v16h a, v16h b, v8f c) {
  // D = A(16x32 f16) * B(32x16 f16) + C(16x16 f32)
  return __builtin_amdgcn_wmma_f32_16x16x32_f16(
      /*neg_a=*/false, a, /*neg_b=*/false, b,
      /*c_mod=*/(short)0, c, /*reuse_a=*/false, /*reuse_b=*/false);
}

__device__ __forceinline__ v8f zero8() {
  v8f z;
#pragma unroll
  for (int i = 0; i < 8; ++i) z[i] = 0.0f;
  return z;
}

// Async copy of 16 bytes global -> LDS (per lane), tracked by ASYNCcnt.
// LDS flat address low 32 bits == LDS byte offset (ISA 10.2 aperture rules).
__device__ __forceinline__ void async_copy16(void* lds_dst, const void* gsrc) {
  unsigned laddr = (unsigned)(unsigned long long)lds_dst;
  unsigned long long gaddr = (unsigned long long)gsrc;
  asm volatile("global_load_async_to_lds_b128 %0, %1, off"
               :: "v"(laddr), "v"(gaddr)
               : "memory");
}

__device__ __forceinline__ void wait_async0() {
  asm volatile("s_wait_asynccnt 0x0" ::: "memory");
}

// Load one A/B fragment (16 halfs) for k = 0..31 relative to rowptr.
// ISA layout: element h -> k = (h/8)*16 + (lane/16)*8 + (h%8).
__device__ __forceinline__ v16h frag_h(const _Float16* rowptr, int lane) {
  const int off = (lane >> 4) << 3;
  v8h lo = *(const v8h*)(rowptr + off);        // k = off .. off+7
  v8h hi = *(const v8h*)(rowptr + 16 + off);   // k = 16+off .. 16+off+7
  v16h r;
#pragma unroll
  for (int i = 0; i < 8; ++i) { r[i] = lo[i]; r[i + 8] = hi[i]; }
  return r;
}

// Same fragment, sourced from f32 memory with inline conversion.
__device__ __forceinline__ v16h frag_f32(const float* rowptr, int lane) {
  const int off = (lane >> 4) << 3;
  v4f a0 = *(const v4f*)(rowptr + off);
  v4f a1 = *(const v4f*)(rowptr + off + 4);
  v4f b0 = *(const v4f*)(rowptr + 16 + off);
  v4f b1 = *(const v4f*)(rowptr + 16 + off + 4);
  v16h r;
#pragma unroll
  for (int i = 0; i < 4; ++i) {
    r[i]      = (_Float16)a0[i];
    r[i + 4]  = (_Float16)a1[i];
    r[i + 8]  = (_Float16)b0[i];
    r[i + 12] = (_Float16)b1[i];
  }
  return r;
}

// ---------------------------------------------------------------------------
// Kernel A: per-head projection  y = x_h @ W_h^T + b_h   (2048 x 64 x 64)
// grid = HEADS*16 blocks of 256 threads; each wave does a 16x64 output tile.
// v_transposed==0: out[h][b][e] (Q,K).  v_transposed==1: out[h][e][b] (V^T).
// ---------------------------------------------------------------------------
__global__ __launch_bounds__(256) void proj_kernel(
    const float* __restrict__ x, const float* __restrict__ W,
    const float* __restrict__ bias, _Float16* __restrict__ out,
    int v_transposed) {
  const int bx   = blockIdx.x;
  const int h    = bx >> 4;
  const int rb   = bx & 15;
  const int lane = threadIdx.x & 31;
  const int wave = threadIdx.x >> 5;
  const int m0   = rb * 128 + wave * 16;

  const float* xrow = x + (size_t)(m0 + (lane & 15)) * DMODEL + h * DHEAD;
  v16h a0 = frag_f32(xrow, lane);       // k = 0..31  (input dim)
  v16h a1 = frag_f32(xrow + 32, lane);  // k = 32..63

  const float* Wh = W + (size_t)h * DHEAD * DHEAD;  // [e][d] row-major
  v8f acc[4];
#pragma unroll
  for (int nt = 0; nt < 4; ++nt) {
    const float* wrow = Wh + (size_t)(nt * 16 + (lane & 15)) * DHEAD;
    v16h b0 = frag_f32(wrow, lane);       // B(k=d, n=e) = W[e][d]
    v16h b1 = frag_f32(wrow + 32, lane);
    v8f c = zero8();
    c = wmma16(a0, b0, c);
    c = wmma16(a1, b1, c);
    acc[nt] = c;
  }

  const float* bh = bias + h * DHEAD;
#pragma unroll
  for (int nt = 0; nt < 4; ++nt) {
#pragma unroll
    for (int r = 0; r < 8; ++r) {
      const int m = m0 + r + ((lane >> 4) << 3);
      const int e = nt * 16 + (lane & 15);
      const float val = acc[nt][r] + bh[e];
      size_t idx;
      if (v_transposed)
        idx = (size_t)h * DHEAD * BATCH + (size_t)e * BATCH + m;
      else
        idx = (size_t)h * BATCH * DHEAD + (size_t)m * DHEAD + e;
      out[idx] = (_Float16)val;
    }
  }
}

// ---------------------------------------------------------------------------
// Kernel B: flash attention per head.  S = Q K^T * 1/8, online softmax,
// O += P V.  grid = HEADS*16 blocks; block handles 128 Q rows of one head.
// K/V stream through double-buffered LDS via async global->LDS copies:
// one barrier per 32-key chunk, copy of chunk i+1 overlaps compute of i.
// ---------------------------------------------------------------------------
__global__ __launch_bounds__(256) void attn_kernel(
    const _Float16* __restrict__ Q, const _Float16* __restrict__ K,
    const _Float16* __restrict__ Vt, _Float16* __restrict__ AO) {
  __shared__ _Float16 ldsK[2][32 * 64];   // [buf][key][e]
  __shared__ _Float16 ldsV[2][64 * 32];   // [buf][e][key]
  __shared__ _Float16 ldsP[8 * 16 * 32];  // per-wave P tile [m][k]

  const int bx   = blockIdx.x;
  const int h    = bx >> 4;
  const int rb   = bx & 15;
  const int lane = threadIdx.x & 31;
  const int wave = threadIdx.x >> 5;
  const int tid  = threadIdx.x;
  const int m0   = rb * 128 + wave * 16;

  const _Float16* Qh = Q  + (size_t)h * BATCH * DHEAD;
  const _Float16* Kh = K  + (size_t)h * BATCH * DHEAD;
  const _Float16* Vh = Vt + (size_t)h * DHEAD * BATCH;  // [e][b]

  const _Float16* qrow = Qh + (size_t)(m0 + (lane & 15)) * DHEAD;
  v16h qa0 = frag_h(qrow, lane);
  v16h qa1 = frag_h(qrow + 32, lane);

  v8f o[4];
  float row_m[8], row_l[8];
#pragma unroll
  for (int nt = 0; nt < 4; ++nt) o[nt] = zero8();
#pragma unroll
  for (int r = 0; r < 8; ++r) { row_m[r] = -1e30f; row_l[r] = 0.0f; }

  const float scale = 0.125f;  // 1/sqrt(64)
  const int kchunk_i = tid * 8;        // 16B of K chunk per thread
  const int ve       = tid >> 2;       // V^T row (e)
  const int vk       = (tid & 3) * 8;  // 16B of that row per thread

  auto issue_chunk = [&](int c0, int buf) {
    async_copy16(&ldsK[buf][kchunk_i], Kh + (size_t)c0 * DHEAD + kchunk_i);
    async_copy16(&ldsV[buf][ve * 32 + vk], Vh + (size_t)ve * BATCH + c0 + vk);
  };

  issue_chunk(0, 0);

  const int NCHUNK = BATCH / 32;
  for (int it = 0; it < NCHUNK; ++it) {
    const int cur = it & 1;
    wait_async0();     // this wave's copies of chunk `it` are in LDS
    __syncthreads();   // ...and everyone else's too; compute(it-1) done
    if (it + 1 < NCHUNK) issue_chunk((it + 1) * 32, cur ^ 1);

    // S tiles: two 16x16 key tiles, K-depth 64 = two chained WMMAs.
    v8f s[2];
#pragma unroll
    for (int t = 0; t < 2; ++t) {
      const _Float16* krow = &ldsK[cur][(t * 16 + (lane & 15)) * 64];
      v16h b0 = frag_h(krow, lane);
      v16h b1 = frag_h(krow + 32, lane);
      v8f c = zero8();
      c = wmma16(qa0, b0, c);
      c = wmma16(qa1, b1, c);
      s[t] = c;
    }

    // Online softmax: row r lives in a fixed 16-lane half -> width-16 shuffles.
    float alpha[8];
#pragma unroll
    for (int r = 0; r < 8; ++r) {
      const float v0 = s[0][r] * scale;
      const float v1 = s[1][r] * scale;
      float mx = fmaxf(v0, v1);
#pragma unroll
      for (int i = 1; i < 16; i <<= 1) mx = fmaxf(mx, __shfl_xor(mx, i, 16));
      const float mn = fmaxf(row_m[r], mx);
      const float al = __expf(row_m[r] - mn);
      const float p0 = __expf(v0 - mn);
      const float p1 = __expf(v1 - mn);
      float sm = p0 + p1;
#pragma unroll
      for (int i = 1; i < 16; i <<= 1) sm += __shfl_xor(sm, i, 16);
      row_l[r] = row_l[r] * al + sm;
      row_m[r] = mn;
      alpha[r] = al;
      s[0][r] = p0;
      s[1][r] = p1;
    }
#pragma unroll
    for (int nt = 0; nt < 4; ++nt)
#pragma unroll
      for (int r = 0; r < 8; ++r) o[nt][r] *= alpha[r];

    // Reshape P (C-layout) -> A-fragment via per-wave LDS tile.
    _Float16* pb = &ldsP[wave * 512];
#pragma unroll
    for (int t = 0; t < 2; ++t)
#pragma unroll
      for (int r = 0; r < 8; ++r) {
        const int m = r + ((lane >> 4) << 3);
        const int k = t * 16 + (lane & 15);
        pb[m * 32 + k] = (_Float16)s[t][r];
      }
    v16h pa = frag_h(&pb[(lane & 15) * 32], lane);

    // O += P[16x32] @ V[32x64]
#pragma unroll
    for (int nt = 0; nt < 4; ++nt) {
      v16h vb = frag_h(&ldsV[cur][(nt * 16 + (lane & 15)) * 32], lane);
      o[nt] = wmma16(pa, vb, o[nt]);
    }
  }

  // Epilogue: normalize and scatter into [b][h*64+e] (f16).
#pragma unroll
  for (int nt = 0; nt < 4; ++nt)
#pragma unroll
    for (int r = 0; r < 8; ++r) {
      const int m = m0 + r + ((lane >> 4) << 3);
      const int e = nt * 16 + (lane & 15);
      const float val = o[nt][r] / row_l[r];
      AO[(size_t)m * DMODEL + h * DHEAD + e] = (_Float16)val;
    }
}

// ---------------------------------------------------------------------------
// Kernel C: out = AO(f16, 2048x2048) @ Wc^T + bc  -> f32.
// grid = 256 blocks; 128x128 tile per block, K-step 32.
// A tile copied async global->LDS (overlaps Wc f32->f16 conversion in VGPRs);
// next Wc k-slab prefetched via global_prefetch.
// ---------------------------------------------------------------------------
__global__ __launch_bounds__(256) void combine_kernel(
    const _Float16* __restrict__ AO, const float* __restrict__ Wc,
    const float* __restrict__ bc, float* __restrict__ out) {
  __shared__ _Float16 ldsA[128 * 32];
  __shared__ _Float16 ldsB[128 * 32];

  const int bx   = blockIdx.x;
  const int mblk = bx >> 4;
  const int nblk = bx & 15;
  const int lane = threadIdx.x & 31;
  const int wave = threadIdx.x >> 5;
  const int tid  = threadIdx.x;

  v8f acc[8];
#pragma unroll
  for (int nt = 0; nt < 8; ++nt) acc[nt] = zero8();

  const int row = tid >> 1;
  const int kk  = (tid & 1) * 16;

  for (int ks = 0; ks < DMODEL; ks += 32) {
    __syncthreads();
    {
      const _Float16* asrc = AO + (size_t)(mblk * 128 + row) * DMODEL + ks + kk;
      async_copy16(&ldsA[row * 32 + kk],     asrc);
      async_copy16(&ldsA[row * 32 + kk + 8], asrc + 8);

      const float* wsrc = Wc + (size_t)(nblk * 128 + row) * DMODEL + ks + kk;
      if (ks + 32 < DMODEL) __builtin_prefetch(wsrc + 32, 0, 1);
      v4f w0 = *(const v4f*)(wsrc);
      v4f w1 = *(const v4f*)(wsrc + 4);
      v4f w2 = *(const v4f*)(wsrc + 8);
      v4f w3 = *(const v4f*)(wsrc + 12);
      v8h h0, h1;
#pragma unroll
      for (int i = 0; i < 4; ++i) {
        h0[i]     = (_Float16)w0[i];
        h0[i + 4] = (_Float16)w1[i];
        h1[i]     = (_Float16)w2[i];
        h1[i + 4] = (_Float16)w3[i];
      }
      *(v8h*)&ldsB[row * 32 + kk]     = h0;
      *(v8h*)&ldsB[row * 32 + kk + 8] = h1;
    }
    wait_async0();
    __syncthreads();

    v16h av = frag_h(&ldsA[(wave * 16 + (lane & 15)) * 32], lane);
#pragma unroll
    for (int nt = 0; nt < 8; ++nt) {
      v16h bv = frag_h(&ldsB[(nt * 16 + (lane & 15)) * 32], lane);
      acc[nt] = wmma16(av, bv, acc[nt]);
    }
  }

#pragma unroll
  for (int nt = 0; nt < 8; ++nt)
#pragma unroll
    for (int r = 0; r < 8; ++r) {
      const int m = mblk * 128 + wave * 16 + r + ((lane >> 4) << 3);
      const int n = nblk * 128 + nt * 16 + (lane & 15);
      out[(size_t)m * DMODEL + n] = acc[nt][r] + bc[n];
    }
}

// ---------------------------------------------------------------------------
extern "C" void kernel_launch(void* const* d_in, const int* in_sizes, int n_in,
                              void* d_out, int out_size, void* d_ws,
                              size_t ws_size, hipStream_t stream) {
  (void)in_sizes; (void)n_in; (void)out_size; (void)ws_size;
  const float* x  = (const float*)d_in[0];
  const float* Wq = (const float*)d_in[1];
  const float* bq = (const float*)d_in[2];
  const float* Wk = (const float*)d_in[3];
  const float* bk = (const float*)d_in[4];
  const float* Wv = (const float*)d_in[5];
  const float* bv = (const float*)d_in[6];
  const float* Wc = (const float*)d_in[7];
  const float* bc = (const float*)d_in[8];
  float* out = (float*)d_out;

  // Workspace layout (f16): Q | K | V^T | AO  = 4x 4M halfs = 32 MB total.
  _Float16* ws = (_Float16*)d_ws;
  const size_t SEG = (size_t)HEADS * BATCH * DHEAD;  // 4M halfs
  _Float16* Qb = ws;
  _Float16* Kb = Qb + SEG;
  _Float16* Vt = Kb + SEG;
  _Float16* AO = Vt + SEG;

  dim3 blk(256);
  proj_kernel<<<HEADS * 16, blk, 0, stream>>>(x, Wq, bq, Qb, 0);
  proj_kernel<<<HEADS * 16, blk, 0, stream>>>(x, Wk, bk, Kb, 0);
  proj_kernel<<<HEADS * 16, blk, 0, stream>>>(x, Wv, bv, Vt, 1);
  attn_kernel<<<HEADS * 16, blk, 0, stream>>>(Qb, Kb, Vt, AO);
  combine_kernel<<<256, blk, 0, stream>>>(AO, Wc, bc, out);
}
